// LSTMDecoder_26001732010241
// MI455X (gfx1250) — matrix operand order repaired and checked by reading, not compiled, not run on hardware
//
#include <hip/hip_runtime.h>
#include <hip/hip_bf16.h>
#include <math.h>

// Problem constants (fixed by reference setup_inputs)
#define V_SZ 32000
#define E_SZ 1024
#define H_SZ 1024
#define L_SZ 2
#define B_SZ 64
#define T_SZ 64

typedef __attribute__((ext_vector_type(16))) __bf16       v16bf;
typedef __attribute__((ext_vector_type(8)))  float        v8f;
typedef __attribute__((ext_vector_type(4)))  unsigned int u32x4;

union ATile { u32x4 q[2]; v16bf v; };

// ---- WMMA A operand loader (contiguous k order used for both A and B, wave32) ----
// A 16x32 bf16: lane L -> m = L%16; element s: K = (L/16)*16 + s
__device__ __forceinline__ v16bf load_a_tile(const __hip_bfloat16* __restrict__ A,
                                             int lda, int m0, int k, int ln, int hf) {
    const __hip_bfloat16* p = A + (size_t)(m0 + ln) * lda + k + hf * 16;
    ATile t;
    t.q[0] = *(const u32x4*)(p);        // K = k+hf*16   .. +7
    t.q[1] = *(const u32x4*)(p + 8);    // K = k+hf*16+8 .. +7
    return t.v;
}

// ---- GEMM: C[64 x N] = A1 @ W1^T (+ A2 @ W2^T) + bias1 (+ bias2), opt ReLU,
//      output f32 (ldc-strided) or bf16.
// Block = 256 thr (8 waves), output tile 64x128.  B operands are staged
// through LDS with double-buffered GLOBAL_LOAD_ASYNC_TO_LDS_B128 (ASYNCcnt),
// stored directly in per-lane WMMA operand order so each consumer lane reads
// one contiguous 32B blob (sequential ds_load_b128, conflict-free), cutting
// L2 weight traffic 4x vs per-wave direct loads.  KT is compile-time so the
// chunk decode is pure shift/mask (no scalar division in the hot loop).
template<int KT, bool DUAL, bool RELU, bool OUTBF16>
__global__ __launch_bounds__(256) void gemm_wmma(
    const __hip_bfloat16* __restrict__ A1, const __hip_bfloat16* __restrict__ W1,
    const __hip_bfloat16* __restrict__ A2, const __hip_bfloat16* __restrict__ W2,
    const float* __restrict__ bias1, const float* __restrict__ bias2,
    float* __restrict__ Cf, __hip_bfloat16* __restrict__ Cb,
    long long ldc)
{
    constexpr int CH1   = KT / 64;                 // K-chunks per matrix (16)
    constexpr int TOTAL = DUAL ? 2 * CH1 : CH1;

    // [buffer][tile(=producer wave)][lane][64 bytes = 2x32B WMMA-B blobs]
    __shared__ alignas(16) unsigned char smem[2][8][32][64];

    const int tid  = threadIdx.x;
    const int lane = tid & 31, wave = tid >> 5;
    const int hf   = lane >> 4, ln = lane & 15;
    const int m0    = (wave & 3) * 16;
    const int ngrp  = wave >> 2;                   // 0..1
    const int nbase = blockIdx.x * 128 + ngrp * 64;

    // Producer role: wave w fills tile w (n = blockbase + w*16 + ln).
    const int nprod = blockIdx.x * 128 + wave * 16 + ln;
    const __hip_bfloat16* w1row = W1 + (size_t)nprod * KT;
    const __hip_bfloat16* w2row = DUAL ? (W2 + (size_t)nprod * KT) : w1row;

    uint32_t lbase[2];
    lbase[0] = (uint32_t)(uintptr_t)&smem[0][wave][lane][0];
    lbase[1] = (uint32_t)(uintptr_t)&smem[1][wave][lane][0];

    // Issue one 64-k chunk (4 async b128 per lane) into LDS buffer `buf`.
    // Lane (n, hf) deposits exactly the two 32B blobs the consumer lanes with
    // the same (ln,hf) will feed to WMMA:
    //   blob0 = W[n][k +    hf*16 .. +15]   -> smem[buf][tile][lane][ 0..31]
    //   blob1 = W[n][k+32 + hf*16 .. +15]   -> smem[buf][tile][lane][32..63]
    auto issue = [&](int buf, int c) {
        const __hip_bfloat16* row = (DUAL && c >= CH1) ? w2row : w1row;
        const __hip_bfloat16* g0  = row + (c & (CH1 - 1)) * 64 + hf * 16;
        uint64_t ga0 = (uint64_t)(uintptr_t)g0;
        uint64_t ga1 = ga0 + 64;               // +32 elements
        uint32_t la0 = lbase[buf];
        uint32_t la1 = la0 + 32;
        // INST_OFFSET is added to BOTH the global and LDS address (ISA §10.7)
        asm volatile(
            "global_load_async_to_lds_b128 %0, %2, off\n\t"
            "global_load_async_to_lds_b128 %0, %2, off offset:16\n\t"
            "global_load_async_to_lds_b128 %1, %3, off\n\t"
            "global_load_async_to_lds_b128 %1, %3, off offset:16"
            :: "v"(la0), "v"(la1), "v"(ga0), "v"(ga1) : "memory");
    };

    v8f acc[4];
    #pragma unroll
    for (int j = 0; j < 4; ++j) {
        const int n = nbase + j * 16 + ln;
        float b = bias1[n];
        if (DUAL) b += bias2[n];
        v8f t = {b, b, b, b, b, b, b, b};
        acc[j] = t;
    }

    auto compute = [&](int c, int p) {
        const __hip_bfloat16* Ap = (DUAL && c >= CH1) ? A2 : A1;
        const int k = (c & (CH1 - 1)) * 64;
        #pragma unroll
        for (int kk = 0; kk < 2; ++kk) {
            const v16bf a = load_a_tile(Ap, KT, m0, k + kk * 32, ln, hf);
            #pragma unroll
            for (int j = 0; j < 4; ++j) {
                const v16bf b = *(const v16bf*)&smem[p][ngrp * 4 + j][lane][kk * 32];
                acc[j] = __builtin_amdgcn_wmma_f32_16x16x32_bf16(
                    false, a, false, b, (short)0, acc[j], false, false);
            }
        }
    };

    issue(0, 0);
    #pragma unroll 2
    for (int c = 0; c < TOTAL - 1; ++c) {          // branch-free steady state
        const int p = c & 1;
        issue(p ^ 1, c + 1);
        asm volatile("s_wait_asynccnt 0x4" ::: "memory");  // chunk c deposited
        __syncthreads();                                   // all tiles visible
        compute(c, p);
        __syncthreads();           // reads done before buffer p refilled at c+2
    }
    // peeled last chunk: nothing left to prefetch, no trailing barrier needed
    asm volatile("s_wait_asynccnt 0x0" ::: "memory");
    __syncthreads();
    compute(TOTAL - 1, (TOTAL - 1) & 1);

    // C/D layout: lane L, VGPR v -> m = m0 + (L/16)*8 + v, n = n0 + L%16
    #pragma unroll
    for (int j = 0; j < 4; ++j) {
        const int n = nbase + j * 16 + ln;
        #pragma unroll
        for (int v = 0; v < 8; ++v) {
            const int m = m0 + hf * 8 + v;
            float val = acc[j][v];
            if (RELU) val = fmaxf(val, 0.0f);
            if (OUTBF16) Cb[(size_t)m * (size_t)ldc + n] = __float2bfloat16(val);
            else         Cf[(size_t)m * (size_t)ldc + n] = val;
        }
    }
}

// ---- LSTM pointwise cell: gates [64,4096] (i|f|g|o), c f32, h -> bf16 ----
__global__ __launch_bounds__(256) void lstm_cell_kernel(
    const float* __restrict__ gates, float* __restrict__ c,
    __hip_bfloat16* __restrict__ h_bf)
{
    const int idx = blockIdx.x * blockDim.x + threadIdx.x;   // 64*1024 threads
    const int b = idx >> 10, hh = idx & 1023;
    const float* g = gates + (size_t)b * (4 * H_SZ);
    const float gi = g[hh], gf = g[H_SZ + hh], gg = g[2 * H_SZ + hh], go = g[3 * H_SZ + hh];
    const float si = 1.0f / (1.0f + __expf(-gi));
    const float sf = 1.0f / (1.0f + __expf(-gf));
    const float so = 1.0f / (1.0f + __expf(-go));
    const float cn = sf * c[idx] + si * tanhf(gg);
    c[idx] = cn;
    h_bf[idx] = __float2bfloat16(so * tanhf(cn));
}

// ---- embed gather (fp32 rows -> bf16 activations) ----
__global__ void gather_embed_kernel(const float* __restrict__ emb,
                                    const int* __restrict__ xt,
                                    __hip_bfloat16* __restrict__ x_bf)
{
    const int b = blockIdx.x;
    const float* src = emb + (size_t)xt[b] * E_SZ;
    for (int e = threadIdx.x; e < E_SZ; e += blockDim.x)
        x_bf[(size_t)b * E_SZ + e] = __float2bfloat16(src[e]);
}

// ---- fp32 -> bf16 weight conversion ----
__global__ void cvt_f32_bf16_kernel(const float* __restrict__ in,
                                    __hip_bfloat16* __restrict__ out, int n)
{
    const int i = blockIdx.x * blockDim.x + threadIdx.x;
    if (i < n) out[i] = __float2bfloat16(in[i]);
}

// ---- state init: h=0 (bf16), c=0, xt=start_tok ----
__global__ void init_state_kernel(__hip_bfloat16* __restrict__ h_bf,
                                  float* __restrict__ c, int* __restrict__ xt,
                                  const int* __restrict__ start_tok)
{
    const int i = blockIdx.x * blockDim.x + threadIdx.x;
    const int n = L_SZ * B_SZ * H_SZ;
    if (i < n) { h_bf[i] = __float2bfloat16(0.0f); c[i] = 0.0f; }
    if (i < B_SZ) xt[i] = start_tok[0];
}

// ---- greedy argmax over V per row; first occurrence on ties (matches jnp) ----
__global__ __launch_bounds__(256) void argmax_kernel(
    const float* __restrict__ scores_t,  // d_out + t*V ; row stride T*V
    int* __restrict__ xt, int* __restrict__ samples_t)  // samples base + t ; stride T
{
    __shared__ float sval[256];
    __shared__ int   sidx[256];
    const int b = blockIdx.x;
    const float* row = scores_t + (size_t)b * ((size_t)T_SZ * V_SZ);
    float best = -INFINITY; int bidx = 0;
    for (int v = threadIdx.x; v < V_SZ; v += 256) {
        const float s = row[v];
        if (s > best) { best = s; bidx = v; }
    }
    sval[threadIdx.x] = best; sidx[threadIdx.x] = bidx;
    __syncthreads();
    for (int off = 128; off > 0; off >>= 1) {
        if (threadIdx.x < off) {
            const float ov = sval[threadIdx.x + off];
            const int   oi = sidx[threadIdx.x + off];
            if (ov > sval[threadIdx.x] ||
                (ov == sval[threadIdx.x] && oi < sidx[threadIdx.x])) {
                sval[threadIdx.x] = ov; sidx[threadIdx.x] = oi;
            }
        }
        __syncthreads();
    }
    if (threadIdx.x == 0) {
        xt[b] = sidx[0];
        samples_t[(size_t)b * T_SZ] = sidx[0];
    }
}

extern "C" void kernel_launch(void* const* d_in, const int* in_sizes, int n_in,
                              void* d_out, int out_size, void* d_ws, size_t ws_size,
                              hipStream_t stream)
{
    const float* emb       = (const float*)d_in[0];
    const float* w_ih      = (const float*)d_in[1];
    const float* w_hh      = (const float*)d_in[2];
    const float* b_ih      = (const float*)d_in[3];
    const float* b_hh      = (const float*)d_in[4];
    const float* o2emb_w   = (const float*)d_in[5];
    const float* o2emb_b   = (const float*)d_in[6];
    const float* score_w   = (const float*)d_in[7];
    const float* score_b   = (const float*)d_in[8];
    const int*   start_tok = (const int*)d_in[11];

    // ---- workspace carving (256B aligned) ----
    char* ws = (char*)d_ws;
    size_t off = 0;
    auto carve = [&](size_t bytes) { char* p = ws + off; off += (bytes + 255) & ~size_t(255); return p; };
    const int N_WIH = L_SZ * 4 * H_SZ * H_SZ;   // 8,388,608
    const int N_WHH = N_WIH;
    const int N_O2W = E_SZ * H_SZ;              // 1,048,576
    const int N_SCW = V_SZ * E_SZ;              // 32,768,000

    __hip_bfloat16* wih_bf = (__hip_bfloat16*)carve((size_t)N_WIH * 2);
    __hip_bfloat16* whh_bf = (__hip_bfloat16*)carve((size_t)N_WHH * 2);
    __hip_bfloat16* o2w_bf = (__hip_bfloat16*)carve((size_t)N_O2W * 2);
    __hip_bfloat16* scw_bf = (__hip_bfloat16*)carve((size_t)N_SCW * 2);
    __hip_bfloat16* x_bf   = (__hip_bfloat16*)carve((size_t)B_SZ * E_SZ * 2);
    __hip_bfloat16* h_bf   = (__hip_bfloat16*)carve((size_t)L_SZ * B_SZ * H_SZ * 2);
    float*          c_st   = (float*)carve((size_t)L_SZ * B_SZ * H_SZ * 4);
    float*          gates  = (float*)carve((size_t)B_SZ * 4 * H_SZ * 4);
    __hip_bfloat16* hid_bf = (__hip_bfloat16*)carve((size_t)B_SZ * E_SZ * 2);
    int*            xt     = (int*)carve((size_t)B_SZ * 4);

    // ---- one-time per launch: quantize weights to bf16 (L2-resident set) ----
    cvt_f32_bf16_kernel<<<(N_WIH + 255) / 256, 256, 0, stream>>>(w_ih,    wih_bf, N_WIH);
    cvt_f32_bf16_kernel<<<(N_WHH + 255) / 256, 256, 0, stream>>>(w_hh,    whh_bf, N_WHH);
    cvt_f32_bf16_kernel<<<(N_O2W + 255) / 256, 256, 0, stream>>>(o2emb_w, o2w_bf, N_O2W);
    cvt_f32_bf16_kernel<<<(N_SCW + 255) / 256, 256, 0, stream>>>(score_w, scw_bf, N_SCW);
    init_state_kernel<<<(L_SZ * B_SZ * H_SZ + 255) / 256, 256, 0, stream>>>(h_bf, c_st, xt, start_tok);

    float* scores  = (float*)d_out;                            // [B, T, V]
    int*   samples = (int*)d_out + (size_t)B_SZ * T_SZ * V_SZ; // [B, T]

    const size_t HB = (size_t)B_SZ * H_SZ;
    const int    W4H = 4 * H_SZ * H_SZ;

    for (int t = 0; t < T_SZ; ++t) {
        gather_embed_kernel<<<B_SZ, 256, 0, stream>>>(emb, xt, x_bf);

        // layer 0: gates = x @ w_ih0^T + h0 @ w_hh0^T + b_ih0 + b_hh0
        gemm_wmma<H_SZ, true, false, false><<<4 * H_SZ / 128, 256, 0, stream>>>(
            x_bf, wih_bf, h_bf, whh_bf, b_ih, b_hh,
            gates, nullptr, 4 * H_SZ);
        lstm_cell_kernel<<<(int)(HB / 256), 256, 0, stream>>>(gates, c_st, h_bf);

        // layer 1: input is h of layer 0
        gemm_wmma<H_SZ, true, false, false><<<4 * H_SZ / 128, 256, 0, stream>>>(
            h_bf, wih_bf + W4H, h_bf + HB, whh_bf + W4H, b_ih + 4 * H_SZ, b_hh + 4 * H_SZ,
            gates, nullptr, 4 * H_SZ);
        lstm_cell_kernel<<<(int)(HB / 256), 256, 0, stream>>>(gates, c_st + HB, h_bf + HB);

        // o2emb: hid = relu(h1 @ o2emb_w^T + b), stored bf16
        gemm_wmma<H_SZ, false, true, true><<<E_SZ / 128, 256, 0, stream>>>(
            h_bf + HB, o2w_bf, nullptr, nullptr, o2emb_b, nullptr,
            nullptr, hid_bf, E_SZ);

        // score: d_out[b][t][:] = hid @ score_w^T + score_b   (ldc = T*V)
        gemm_wmma<E_SZ, false, false, false><<<V_SZ / 128, 256, 0, stream>>>(
            hid_bf, scw_bf, nullptr, nullptr, score_b, nullptr,
            scores + (size_t)t * V_SZ, nullptr, (long long)T_SZ * V_SZ);

        // greedy token + samples output
        argmax_kernel<<<B_SZ, 256, 0, stream>>>(scores + (size_t)t * V_SZ, xt, samples + t);
    }
}
